// MSSAM_88862873354712
// MI455X (gfx1250) — compile-verified, matrix operand order checked by reading, not run on hardware
//
#include <hip/hip_runtime.h>
#include <math.h>

typedef __bf16 bf16_t;
typedef __attribute__((ext_vector_type(16))) __bf16 v16bf;
typedef __attribute__((ext_vector_type(8)))  float  v8f;

#define WMMA_BF16(a, b, c) \
  __builtin_amdgcn_wmma_f32_16x16x32_bf16(false, (a), false, (b), (short)0, (c), false, false)

__device__ __forceinline__ float fast_sigmoid(float x) {
  return __builtin_amdgcn_rcpf(1.0f + __expf(-x));   // v_exp_f32 + v_rcp_f32
}

// ---------------------------------------------------------------------------
// Weight transpose + f32->bf16 convert: wT[c][k] = w[k][c]   (w is [K,256])
// ---------------------------------------------------------------------------
__global__ void wprep_kernel(const float* __restrict__ w, bf16_t* __restrict__ wT, int K) {
  int gid = blockIdx.x * blockDim.x + threadIdx.x;
  int total = K * 256;
  if (gid >= total) return;
  int c = gid / K;
  int k = gid - c * K;
  wT[gid] = (bf16_t)w[k * 256 + c];
}

// ---------------------------------------------------------------------------
// Projection GEMM: out[r,c] = sum_k fin[r,k]*w[k,c] + bias[c]
// wT bf16 [256,K] staged into LDS (whole for K<=512, two halves for K=1024)
// and shared by all 8 waves; B fragments become ds_load_b128 w/ imm offsets.
// ---------------------------------------------------------------------------
template <int K>
__global__ __launch_bounds__(256) void proj_kernel(
    const float* __restrict__ fin, const bf16_t* __restrict__ wT,
    const float* __restrict__ bias, bf16_t* __restrict__ xout,
    float* __restrict__ pout) {
  constexpr int KS = (K > 512) ? 512 : K;       // staged K-columns per pass
  constexpr int NSTAGE = K / KS;
  __shared__ __align__(16) bf16_t wlds[256 * KS];

  const int tid  = threadIdx.x;
  const int lane = tid & 31;
  const int wid  = tid >> 5;
  const int rbase = (blockIdx.x * 8 + wid) * 16;
  const int l15  = lane & 15;
  const int hi16 = (lane >> 4) << 4;
  const int hi8  = (lane >> 4) << 3;

  v8f acc[16];
#pragma unroll
  for (int c = 0; c < 16; ++c) { v8f z = {}; acc[c] = z; }

  for (int s = 0; s < NSTAGE; ++s) {
    __syncthreads();                             // guard previous consumption
    {                                            // stage wT[:, s*KS .. s*KS+KS)
      const uint4* src = (const uint4*)(wT + (size_t)tid * K + s * KS);
      uint4* dst = (uint4*)(wlds + (size_t)tid * KS);
#pragma unroll
      for (int i = 0; i < KS / 8; ++i) dst[i] = src[i];
    }
    __syncthreads();

    const float*  aBase = fin + (size_t)(rbase + l15) * K + s * KS + hi16;
    const bf16_t* bBase = wlds + (size_t)l15 * KS + hi16;

    for (int kc = 0; kc < (KS >> 5); ++kc) {
      const float* ap = aBase + kc * 32;
      __builtin_prefetch(ap + 32, 0, 3);
      v16bf a;
#pragma unroll
      for (int i = 0; i < 16; ++i) a[i] = (bf16_t)ap[i];

      const bf16_t* bp = bBase + kc * 32;
      v16bf bcur = *(const v16bf*)bp;            // c = 0
#pragma unroll
      for (int c = 0; c < 16; ++c) {
        v16bf bnext = bcur;
        if (c < 15) bnext = *(const v16bf*)(bp + (size_t)((c + 1) * 16 * KS));
        acc[c] = WMMA_BF16(a, bcur, acc[c]);
        bcur = bnext;
      }
    }
  }

  const size_t base0 = (size_t)(rbase + hi8) * 256 + l15;
#pragma unroll
  for (int c = 0; c < 16; ++c) {
    float bv = bias[c * 16 + l15];
    bf16_t* xrow = xout + base0 + c * 16;
    float*  prow = pout ? pout + base0 + c * 16 : nullptr;
#pragma unroll
    for (int v = 0; v < 8; ++v) {
      float val = acc[c][v] + bv;
      xrow[(size_t)v * 256] = (bf16_t)val;
      if (prow) prow[(size_t)v * 256] = val;
    }
  }
}

// ---------------------------------------------------------------------------
// x1 [16,1024,256] bf16 -> x1T [16,256,1024] bf16
// ---------------------------------------------------------------------------
__global__ void xt_kernel(const bf16_t* __restrict__ x, bf16_t* __restrict__ xT) {
  int gid = blockIdx.x * blockDim.x + threadIdx.x;
  if (gid >= 16 * 1024 * 256) return;
  int b = gid >> 18;
  int rem = gid & 262143;
  int m = rem >> 8;
  int c = rem & 255;
  xT[((size_t)b << 18) + ((size_t)c << 10) + m] = x[gid];
}

// ---------------------------------------------------------------------------
// Gram: att[b] = x[b] @ x[b]^T,  x bf16 [16,N,256], att f32 [16,N,N].
// ---------------------------------------------------------------------------
template <int N>
__global__ __launch_bounds__(256) void gram_kernel(
    const bf16_t* __restrict__ x, float* __restrict__ att) {
  const int lane = threadIdx.x & 31;
  const int wid  = threadIdx.x >> 5;
  constexpr int tilesPerBatch = N >> 4;
  const int gw = blockIdx.x * 8 + wid;
  const int batch = gw / tilesPerBatch;
  const int t = gw - batch * tilesPerBatch;
  if (batch >= 16) return;
  const int l15 = lane & 15, hi16 = (lane >> 4) << 4, hi8 = (lane >> 4) << 3;
  const bf16_t* xb = x + (size_t)batch * N * 256;
  float* ab = att + (size_t)batch * N * N;
  const int rbase = t * 16;

  v16bf A[8];
#pragma unroll
  for (int k = 0; k < 8; ++k)
    A[k] = *(const v16bf*)(xb + (size_t)(rbase + l15) * 256 + k * 32 + hi16);

  for (int mt = 0; mt < tilesPerBatch; ++mt) {
    const bf16_t* bp = xb + (size_t)(mt * 16 + l15) * 256 + hi16;
    v8f acc = {};
    v16bf bcur = *(const v16bf*)bp;
#pragma unroll
    for (int k = 0; k < 8; ++k) {
      v16bf bnext = bcur;
      if (k < 7) bnext = *(const v16bf*)(bp + (k + 1) * 32);
      acc = WMMA_BF16(A[k], bcur, acc);
      bcur = bnext;
    }
    float* arow = ab + (size_t)(rbase + hi8) * N + mt * 16 + l15;
#pragma unroll
    for (int v = 0; v < 8; ++v) arow[(size_t)v * N] = acc[v];
  }
}

// ---------------------------------------------------------------------------
// attR[b,n,m] = a2*bilerp(att2, 256->1024) + a3*bilerp(att3, 64->1024)
// Legacy TF resize: src = dst*in/out, lo=floor, hi=min(lo+1,in-1).
// ---------------------------------------------------------------------------
__global__ void resize_kernel(const float* __restrict__ att2, const float* __restrict__ att3,
                              const float* __restrict__ a2p, const float* __restrict__ a3p,
                              float* __restrict__ attR) {
  size_t gid = (size_t)blockIdx.x * blockDim.x + threadIdx.x;
  int b = (int)(gid >> 20);
  if (b >= 16) return;
  int rem = (int)(gid & 1048575);
  int n = rem >> 10, m = rem & 1023;
  const float a2 = *a2p, a3 = *a3p;

  const float* A2 = att2 + ((size_t)b << 16);
  int n0 = n >> 2; float fn = (float)(n & 3) * 0.25f;   int n1 = n0 + 1 < 256 ? n0 + 1 : 255;
  int m0 = m >> 2; float fm = (float)(m & 3) * 0.25f;   int m1 = m0 + 1 < 256 ? m0 + 1 : 255;
  float v2 = (A2[n0 * 256 + m0] * (1.f - fm) + A2[n0 * 256 + m1] * fm) * (1.f - fn)
           + (A2[n1 * 256 + m0] * (1.f - fm) + A2[n1 * 256 + m1] * fm) * fn;

  const float* A3 = att3 + ((size_t)b << 12);
  int p0 = n >> 4; float gn = (float)(n & 15) * 0.0625f; int p1 = p0 + 1 < 64 ? p0 + 1 : 63;
  int q0 = m >> 4; float gm = (float)(m & 15) * 0.0625f; int q1 = q0 + 1 < 64 ? q0 + 1 : 63;
  float v3 = (A3[p0 * 64 + q0] * (1.f - gm) + A3[p0 * 64 + q1] * gm) * (1.f - gn)
           + (A3[p1 * 64 + q0] * (1.f - gm) + A3[p1 * 64 + q1] * gm) * gn;

  attR[gid] = a2 * v2 + a3 * v3;
}

// ---------------------------------------------------------------------------
// Fused: att1 tile -> sigmoid(a1*att1 + attR) -> (bf16, LDS relayout) ->
//        out += att @ x1 ; epilogue out += p1.
// x1/x1T m-panels staged in double-buffered LDS, shared by all 8 waves
// (all waves of a block provably work on the same batch).
// ---------------------------------------------------------------------------
__global__ __launch_bounds__(256) void fused_kernel(
    const bf16_t* __restrict__ x1, const bf16_t* __restrict__ x1T,
    const float* __restrict__ p1, const float* __restrict__ attR,
    const float* __restrict__ a1p, float* __restrict__ out) {
  __shared__ __align__(16) bf16_t sAtt[2][32 * 256];   // x1 rows  m0..m0+31   (16 KB x2)
  __shared__ __align__(16) bf16_t sOut[2][256 * 32];   // x1T cols m0..m0+31   (16 KB x2)
  __shared__ __align__(32) bf16_t stage[8][16 * 32];   // per-wave D->A relayout (8 KB)

  const int tid  = threadIdx.x;
  const int lane = tid & 31;
  const int wid  = tid >> 5;
  const int gw = blockIdx.x * 8 + wid;        // 0..1023
  const int batch = gw >> 6;                  // uniform across the block
  const int rbase = (gw & 63) * 16;
  const int l15 = lane & 15, hi16 = (lane >> 4) << 4, hi8 = (lane >> 4) << 3;
  const bf16_t* xb  = x1  + ((size_t)batch << 18);
  const bf16_t* xTb = x1T + ((size_t)batch << 18);
  const float*  aRb = attR + ((size_t)batch << 20);
  const float a1 = *a1p;
  bf16_t* st = stage[wid];

  auto copyPanels = [&](int buf, int m0) {
    // x1 panel: 32 rows x 256 halfs, contiguous 16 KB
    {
      const uint4* src = (const uint4*)(xb + (size_t)m0 * 256);
      uint4* dst = (uint4*)sAtt[buf];
#pragma unroll
      for (int i = 0; i < 4; ++i) dst[tid + 256 * i] = src[tid + 256 * i];
    }
    // x1T panel: 256 rows x 32 halfs (64 B per row), row stride 1024 halfs
    {
      const uint4* src = (const uint4*)(xTb + (size_t)tid * 1024 + m0);
      uint4* dst = (uint4*)(sOut[buf] + tid * 32);
#pragma unroll
      for (int i = 0; i < 4; ++i) dst[i] = src[i];
    }
  };

  v16bf A1[8];
#pragma unroll
  for (int k = 0; k < 8; ++k)
    A1[k] = *(const v16bf*)(xb + (size_t)(rbase + l15) * 256 + k * 32 + hi16);

  v8f oacc[16];
#pragma unroll
  for (int c = 0; c < 16; ++c) { v8f z = {}; oacc[c] = z; }

  copyPanels(0, 0);
  __syncthreads();

  for (int m0 = 0; m0 < 1024; m0 += 32) {
    const int buf = (m0 >> 5) & 1;
    if (m0 + 32 < 1024) copyPanels(buf ^ 1, m0 + 32);   // prefetch next panels

    // ---- att1 tiles + sigmoid -> bf16 relayout via per-wave LDS ----
    const bf16_t* pAtt = sAtt[buf] + (size_t)l15 * 256 + hi16;
#pragma unroll
    for (int sub = 0; sub < 2; ++sub) {
      const bf16_t* bp = pAtt + sub * 16 * 256;
      v8f acc = {};
      v16bf bcur = *(const v16bf*)bp;
#pragma unroll
      for (int k = 0; k < 8; ++k) {
        v16bf bnext = bcur;
        if (k < 7) bnext = *(const v16bf*)(bp + (k + 1) * 32);
        acc = WMMA_BF16(A1[k], bcur, acc);
        bcur = bnext;
      }
      const int mcol = m0 + sub * 16 + l15;
      const float* aRrow = aRb + (size_t)(rbase + hi8) * 1024 + mcol;
#pragma unroll
      for (int v = 0; v < 8; ++v) {
        float val = fmaf(a1, acc[v], aRrow[(size_t)v * 1024]);
        st[(v + hi8) * 32 + sub * 16 + l15] = (bf16_t)fast_sigmoid(val);
      }
    }
    __asm__ volatile("s_wait_dscnt 0" ::: "memory");      // stores -> loads (same wave)
    v16bf Aa = *(const v16bf*)(st + l15 * 32 + hi16);     // A-fragment relayout

    // ---- out += att @ x1 over this 32-K chunk ----
    const bf16_t* bp = sOut[buf] + (size_t)l15 * 32 + hi16;
    v16bf bcur = *(const v16bf*)bp;
#pragma unroll
    for (int c = 0; c < 16; ++c) {
      v16bf bnext = bcur;
      if (c < 15) bnext = *(const v16bf*)(bp + (size_t)((c + 1) * 16 * 32));
      oacc[c] = WMMA_BF16(Aa, bcur, oacc[c]);
      bcur = bnext;
    }
    __syncthreads();   // all waves done with buf; next iter's writes are safe
  }

  const size_t base0 = ((size_t)batch << 18) + (size_t)(rbase + hi8) * 256 + l15;
#pragma unroll
  for (int c = 0; c < 16; ++c) {
    const float* prow = p1 + base0 + c * 16;
    float*       orow = out + base0 + c * 16;
#pragma unroll
    for (int v = 0; v < 8; ++v)
      orow[(size_t)v * 256] = oacc[c][v] + prow[(size_t)v * 256];
  }
}

// ---------------------------------------------------------------------------
extern "C" void kernel_launch(void* const* d_in, const int* in_sizes, int n_in,
                              void* d_out, int out_size, void* d_ws, size_t ws_size,
                              hipStream_t stream) {
  (void)in_sizes; (void)n_in; (void)out_size; (void)ws_size;
  const float* f1 = (const float*)d_in[0];
  const float* f2 = (const float*)d_in[1];
  const float* f3 = (const float*)d_in[2];
  const float* w1 = (const float*)d_in[3];
  const float* b1 = (const float*)d_in[4];
  const float* w2 = (const float*)d_in[5];
  const float* b2 = (const float*)d_in[6];
  const float* w3 = (const float*)d_in[7];
  const float* b3 = (const float*)d_in[8];
  const float* a1 = (const float*)d_in[9];
  const float* a2 = (const float*)d_in[10];
  const float* a3 = (const float*)d_in[11];
  float* out = (float*)d_out;

  char* ws = (char*)d_ws;
  size_t off = 0;
  auto bump = [&](size_t bytes) -> void* {
    void* p = ws + off;
    off += (bytes + 255) & ~(size_t)255;
    return p;
  };
  bf16_t* x1   = (bf16_t*)bump(16ull * 1024 * 256 * 2);   //  8 MB
  bf16_t* x1T  = (bf16_t*)bump(16ull * 1024 * 256 * 2);   //  8 MB
  float*  p1   = (float*) bump(16ull * 1024 * 256 * 4);   // 16 MB
  bf16_t* x2   = (bf16_t*)bump(16ull * 256 * 256 * 2);    //  2 MB
  bf16_t* x3   = (bf16_t*)bump(16ull * 64 * 256 * 2);     // .5 MB
  bf16_t* w1T  = (bf16_t*)bump(256ull * 256 * 2);
  bf16_t* w2T  = (bf16_t*)bump(256ull * 512 * 2);
  bf16_t* w3T  = (bf16_t*)bump(256ull * 1024 * 2);
  float*  att2 = (float*) bump(16ull * 256 * 256 * 4);    //  4 MB
  float*  att3 = (float*) bump(16ull * 64 * 64 * 4);
  float*  attR = (float*) bump(16ull * 1024 * 1024 * 4);  // 64 MB

  // 1) weights: transpose + bf16
  wprep_kernel<<<(256 * 256 + 255) / 256, 256, 0, stream>>>(w1, w1T, 256);
  wprep_kernel<<<(512 * 256 + 255) / 256, 256, 0, stream>>>(w2, w2T, 512);
  wprep_kernel<<<(1024 * 256 + 255) / 256, 256, 0, stream>>>(w3, w3T, 1024);

  // 2) projections (WMMA, weights staged in LDS)
  proj_kernel<256><<<128, 256, 0, stream>>>(f1, w1T, b1, x1, p1);
  proj_kernel<512><<<32,  256, 0, stream>>>(f2, w2T, b2, x2, nullptr);
  proj_kernel<1024><<<8,  256, 0, stream>>>(f3, w3T, b3, x3, nullptr);

  // 3) x1 transpose for the output GEMM's B-fragments
  xt_kernel<<<(16 * 1024 * 256) / 256, 256, 0, stream>>>(x1, x1T);

  // 4) Gram matrices (WMMA)
  gram_kernel<256><<<32, 256, 0, stream>>>(x2, att2);
  gram_kernel<64><<<8,   256, 0, stream>>>(x3, att3);

  // 5) bilinear resize + scale-combine
  resize_kernel<<<(16 * 1024 * 1024) / 256, 256, 0, stream>>>(att2, att3, a2, a3, attR);

  // 6) fused att1 + sigmoid + att@x1 + p1 (WMMA, LDS-staged panels)
  fused_kernel<<<128, 256, 0, stream>>>(x1, x1T, p1, attR, a1, out);
}
